// PrimitiveAttentiveAtlasEncoder_85804856640021
// MI455X (gfx1250) — matrix operand order repaired and checked by reading, not compiled, not run on hardware
//
#include <hip/hip_runtime.h>

// Problem constants (from reference): B=16384, D=64, K=128, NC=64, F=256
#define B_TOT   16384
#define D_      64
#define K_      128
#define NC_     64
#define F_      256
#define ROWS_PER_BLOCK 128
#define THREADS 256

typedef unsigned short u16;
typedef unsigned int   u32;

typedef __attribute__((ext_vector_type(16))) __bf16 v16bf;
typedef __attribute__((ext_vector_type(8)))  float  v8f;

union FragAB {
  v16bf bf;
  u16   u[16];
  u32   d[8];
  uint4 q4[2];
};
union FragC {
  v8f   f;
  float x[8];
};

__device__ __forceinline__ u16 f2bf(float f) {
  u32 u = __builtin_bit_cast(u32, f);
  u += 0x7FFFu + ((u >> 16) & 1u);      // round-to-nearest-even
  return (u16)(u >> 16);
}
__device__ __forceinline__ float bf2f(u16 h) {
  u32 u = ((u32)h) << 16;
  return __builtin_bit_cast(float, u);
}
__device__ __forceinline__ u32 pack2(float lo, float hi) {
  return (u32)f2bf(lo) | ((u32)f2bf(hi) << 16);
}
__device__ __forceinline__ v8f wmma_bf16(v16bf a, v16bf b, v8f c) {
  // D(f32 16x16) = A(bf16 16x32) * B(bf16 32x16) + C
  return __builtin_amdgcn_wmma_f32_16x16x32_bf16(false, a, false, b, (short)0, c,
                                                 false, false);
}

// A-fragment (16x32 bf16) from LDS, row-major [row][k] with given stride.
// lanes 0-15: K = k0+0..7 (V0-3), k0+16..23 (V4-7); lanes 16-31: +8.
__device__ __forceinline__ FragAB loadA(const u16* base, int stride, int row,
                                        int k0, int g) {
  FragAB f;
  int o = row * stride + k0 + g * 8;
  f.q4[0] = *(const uint4*)(base + o);
  f.q4[1] = *(const uint4*)(base + o + 16);
  return f;
}
// B-fragment (32x16 bf16) from LDS where the weight is stored row-major as
// [n][t] (n = output column, t = reduction). lanes 0-15: t = k0+0..7, k0+16..23; lanes 16-31: +8.
__device__ __forceinline__ FragAB loadB(const u16* base, int stride, int col,
                                        int k0, int g) {
  FragAB f;
  int o = col * stride + k0 + g * 8;
  f.q4[0] = *(const uint4*)(base + o);
  f.q4[1] = *(const uint4*)(base + o + 16);
  return f;
}

// ---- LDS layout (bytes) -----------------------------------------------------
#define OFF_Z      0        // f32  [128][68]  (padded: bank-conflict-free col reads)
#define OFF_ZBF    34816    // bf16 [128][64]
#define OFF_FEAT   51200    // bf16 [128][256]   -- aliased later by scores f32 [128][65]
#define OFF_QZW    116736   // bf16 [128][64]
#define OFF_QFW    133120   // bf16 [128][256]
#define OFF_CQ     198656   // bf16 [64][128]
#define OFF_GS     215040   // bf16 [128][64]  gamma a-slab
#define OFF_Q      231424   // bf16 [128][128] post-bias q (A-operand of q@cq^T)
#define OFF_ZSQ    264192   // f32 [128]
#define OFF_ITAU   264704   // f32 [128]
#define OFF_ILAM   265216   // f32 [128]
#define OFF_QB     265728   // f32 [128]
#define OFF_CSQ    266240   // f32 [64]
#define SMEM_BYTES 266496   // 260.25 KB < 320 KB WGP LDS

__global__ void __launch_bounds__(THREADS, 1)
PrimitiveAttentiveAtlasEncoder_85804856640021_kernel(
    const float* __restrict__ z, const float* __restrict__ feat,
    const float* __restrict__ qzw, const float* __restrict__ qzb,
    const float* __restrict__ qfw, const float* __restrict__ qfb,
    const float* __restrict__ qgamma, const float* __restrict__ cq,
    float* __restrict__ out) {
  extern __shared__ char smem[];
  float* s_z    = (float*)(smem + OFF_Z);
  u16*   s_zbf  = (u16*)(smem + OFF_ZBF);
  u16*   s_feat = (u16*)(smem + OFF_FEAT);
  float* s_sc   = (float*)(smem + OFF_FEAT);   // alias, used after feat consumed
  u16*   s_qzw  = (u16*)(smem + OFF_QZW);
  u16*   s_qfw  = (u16*)(smem + OFF_QFW);
  u16*   s_cq   = (u16*)(smem + OFF_CQ);
  u16*   s_gs   = (u16*)(smem + OFF_GS);
  u16*   s_q    = (u16*)(smem + OFF_Q);
  float* s_zsq  = (float*)(smem + OFF_ZSQ);
  float* s_itau = (float*)(smem + OFF_ITAU);
  float* s_ilam = (float*)(smem + OFF_ILAM);
  float* s_qb   = (float*)(smem + OFF_QB);
  float* s_csq  = (float*)(smem + OFF_CSQ);

  const int tid  = threadIdx.x;
  const int row0 = blockIdx.x * ROWS_PER_BLOCK;

  // ---------------- Stage 0: cooperative staging into LDS -------------------
  {
    const float4* zg = (const float4*)(z + (size_t)row0 * D_);
    for (int i = tid; i < ROWS_PER_BLOCK * D_ / 4; i += THREADS) {
      float4 v = zg[i];
      int r = i >> 4, c4 = i & 15;
      *(float4*)(s_z + r * 68 + c4 * 4) = v;
      *(uint2*)(s_zbf + i * 4) = make_uint2(pack2(v.x, v.y), pack2(v.z, v.w));
    }
    const float4* fg = (const float4*)(feat + (size_t)row0 * F_);
    for (int i = tid; i < ROWS_PER_BLOCK * F_ / 4; i += THREADS) {
      float4 v = fg[i];
      *(uint2*)(s_feat + i * 4) = make_uint2(pack2(v.x, v.y), pack2(v.z, v.w));
    }
    const float4* wg = (const float4*)qzw;
    for (int i = tid; i < K_ * D_ / 4; i += THREADS) {
      float4 v = wg[i];
      *(uint2*)(s_qzw + i * 4) = make_uint2(pack2(v.x, v.y), pack2(v.z, v.w));
    }
    const float4* fw = (const float4*)qfw;
    for (int i = tid; i < K_ * F_ / 4; i += THREADS) {
      float4 v = fw[i];
      *(uint2*)(s_qfw + i * 4) = make_uint2(pack2(v.x, v.y), pack2(v.z, v.w));
    }
    const float4* cg = (const float4*)cq;
    for (int i = tid; i < NC_ * K_ / 4; i += THREADS) {
      float4 v = cg[i];
      *(uint2*)(s_cq + i * 4) = make_uint2(pack2(v.x, v.y), pack2(v.z, v.w));
    }
  }
  __syncthreads();

  // Per-row scalars: z_sq, 1/tau, 1/lam, fused bias; per-center c_sq.
  if (tid < ROWS_PER_BLOCK) {
    const float* zr = s_z + tid * 68;
    float acc = 0.f;
    for (int d = 0; d < D_; ++d) acc += zr[d] * zr[d];
    s_zsq[tid] = acc;
    float tau = 11.313708499f /*sqrt(K)*/ * fmaxf(1.f - acc, 0.001f) * 0.5f;
    tau = fmaxf(tau, 0.01f);
    s_itau[tid] = 1.f / tau;
    float r2 = fminf(acc, 1.f - 0.001f);
    s_ilam[tid] = (1.f - r2 + 0.001f) * 0.5f;   // == 1/lam
    s_qb[tid] = qzb[tid] + qfb[tid];
  } else if (tid < 192) {
    int n = tid - 128;
    const float* cp = cq + n * K_;
    float acc = 0.f;
    for (int d = 0; d < D_; ++d) acc += cp[d] * cp[d];
    s_csq[n] = acc;
  }
  __syncthreads();

  // ---------------- Per-wave WMMA pipeline ----------------------------------
  const int lane = tid & 31;
  const int wave = tid >> 5;
  const int g    = lane >> 4;       // lane half
  const int g8   = g << 3;
  const int lr   = lane & 15;
  const int wrow = wave * 16;       // local row base of this wave's 16-row tile
  const int arow = wrow + lr;       // A-operand row for this lane

  FragAB zA[2];
#pragma unroll
  for (int ks = 0; ks < 2; ++ks) zA[ks] = loadA(s_zbf, D_, arow, ks * 32, g);

  // (a) z @ centers^T  -> sc[4] tiles (16 x 64 f32)
  FragC sc[4];
#pragma unroll
  for (int t = 0; t < 4; ++t)
#pragma unroll
    for (int v = 0; v < 8; ++v) sc[t].x[v] = 0.f;
#pragma unroll
  for (int tn = 0; tn < 4; ++tn)
#pragma unroll
    for (int ks = 0; ks < 2; ++ks) {
      FragAB fb = loadB(s_cq, K_, tn * 16 + lr, ks * 32, g);
      sc[tn].f = wmma_bf16(zA[ks].bf, fb.bf, sc[tn].f);
    }

  // (b) q accumulators (16 x 128 f32)
  FragC qacc[8];
#pragma unroll
  for (int t = 0; t < 8; ++t)
#pragma unroll
    for (int v = 0; v < 8; ++v) qacc[t].x[v] = 0.f;

  // z @ q_z_w^T
#pragma unroll
  for (int ks = 0; ks < 2; ++ks)
#pragma unroll
    for (int nt = 0; nt < 8; ++nt) {
      FragAB fb = loadB(s_qzw, D_, nt * 16 + lr, ks * 32, g);
      qacc[nt].f = wmma_bf16(zA[ks].bf, fb.bf, qacc[nt].f);
    }

  // features @ q_feat_w^T
  for (int ks = 0; ks < 8; ++ks) {
    FragAB fa = loadA(s_feat, F_, arow, ks * 32, g);
#pragma unroll
    for (int nt = 0; nt < 8; ++nt) {
      FragAB fb = loadB(s_qfw, F_, nt * 16 + lr, ks * 32, g);
      qacc[nt].f = wmma_bf16(fa.bf, fb.bf, qacc[nt].f);
    }
  }

  // gamma term: q2[b,k] = <vec(z_b z_b^T), vec(G_k)>.  Reduction chunk
  // c = a*64 + j: A-frag = zA[j-half] scaled per-row by z[b,a]; B-frag = G slab.
  for (int a = 0; a < D_; ++a) {
    __syncthreads();                      // previous slab fully consumed
    {
      int k  = tid >> 1;
      int j0 = (tid & 1) * 32;
      const float4* gp = (const float4*)(qgamma + ((size_t)k * D_ + a) * D_ + j0);
      u16* dst = s_gs + k * D_ + j0;
#pragma unroll
      for (int u = 0; u < 8; ++u) {
        float4 v = gp[u];
        *(uint2*)(dst + u * 4) = make_uint2(pack2(v.x, v.y), pack2(v.z, v.w));
      }
    }
    __syncthreads();                      // slab visible to all 8 waves
    float sf = s_z[arow * 68 + a];        // per-row scalar z[b, a]
#pragma unroll
    for (int h = 0; h < 2; ++h) {
      // sa = zA[h] * sf elementwise (bf16), done per packed dword in fp32.
      FragAB sa;
#pragma unroll
      for (int e = 0; e < 8; ++e) {
        u32 w = zA[h].d[e];
        float lo = __builtin_bit_cast(float, w << 16);
        float hi = __builtin_bit_cast(float, w & 0xFFFF0000u);
        sa.d[e] = pack2(lo * sf, hi * sf);
      }
#pragma unroll
      for (int nt = 0; nt < 8; ++nt) {
        FragAB fb = loadB(s_gs, D_, nt * 16 + lr, h * 32, g);
        qacc[nt].f = wmma_bf16(sa.bf, fb.bf, qacc[nt].f);
      }
    }
  }

  // (c) bias, then park q as bf16 in LDS (A-layout source for q @ cq^T)
#pragma unroll
  for (int nt = 0; nt < 8; ++nt) {
    float bias = s_qb[nt * 16 + lr];
#pragma unroll
    for (int v = 0; v < 8; ++v) {
      float val = qacc[nt].x[v] + bias;
      s_q[(wrow + v + g8) * K_ + nt * 16 + lr] = f2bf(val);
    }
  }
  // wave-private LDS region: same-wave DS ordering, no barrier needed

  // (d) q @ chart_queries^T -> qc[4] tiles
  FragC qc[4];
#pragma unroll
  for (int t = 0; t < 4; ++t)
#pragma unroll
    for (int v = 0; v < 8; ++v) qc[t].x[v] = 0.f;
#pragma unroll
  for (int ks = 0; ks < 4; ++ks) {
    FragAB qa = loadA(s_q, K_, arow, ks * 32, g);
#pragma unroll
    for (int tn = 0; tn < 4; ++tn) {
      FragAB fb = loadB(s_cq, K_, tn * 16 + lr, ks * 32, g);
      qc[tn].f = wmma_bf16(qa.bf, fb.bf, qc[tn].f);
    }
  }

  // (e) fp32 epilogue: hyperbolic distance + tangent-query score
  float zsqv[8], itv[8], ilv[8];
#pragma unroll
  for (int v = 0; v < 8; ++v) {
    int r = wrow + v + g8;
    zsqv[v] = s_zsq[r];
    itv[v]  = s_itau[r];
    ilv[v]  = s_ilam[r];
  }
  const float invsqrtK = 0.08838834764831845f;
#pragma unroll
  for (int tn = 0; tn < 4; ++tn) {
    int col = tn * 16 + lr;
    float csq = s_csq[col];
#pragma unroll
    for (int v = 0; v < 8; ++v) {
      float dot   = sc[tn].x[v];
      float dsq   = zsqv[v] + csq - 2.0f * dot;
      float denom = (1.0f - zsqv[v]) * (1.0f - csq);
      float arg   = 1.0f + 2.0f * dsq / (denom + 0.001f);
      arg = fmaxf(arg, 1.001f);
      float dist = __logf(arg + __builtin_sqrtf(arg * arg - 1.0f));  // arccosh
      float s = -dist * itv[v] + qc[tn].x[v] * ilv[v] * invsqrtK;
      s_sc[(wrow + v + g8) * 65 + col] = s;    // stride 65: conflict-free rows
    }
  }
  __syncthreads();

  // ---------------- Softmax + argmax, one row per thread --------------------
  if (tid < ROWS_PER_BLOCK) {
    float* srow = s_sc + tid * 65;
    float m = -1e30f;
    int am = 0;
    for (int n = 0; n < NC_; ++n) {
      float v = srow[n];
      if (v > m) { m = v; am = n; }
    }
    float sum = 0.f;
    for (int n = 0; n < NC_; ++n) {
      float e = __expf(srow[n] - m);
      srow[n] = e;
      sum += e;
    }
    float inv = 1.0f / sum;
    size_t gr = (size_t)row0 + tid;
    float* op = out + gr * NC_;
    for (int n = 0; n < NC_; ++n) op[n] = srow[n] * inv;
    out[(size_t)B_TOT * NC_ + gr] = (float)am;   // k_chart
  }
}

extern "C" void kernel_launch(void* const* d_in, const int* in_sizes, int n_in,
                              void* d_out, int out_size, void* d_ws, size_t ws_size,
                              hipStream_t stream) {
  (void)in_sizes; (void)n_in; (void)d_ws; (void)ws_size; (void)out_size;
  const float* z        = (const float*)d_in[0];
  const float* features = (const float*)d_in[1];
  const float* q_z_w    = (const float*)d_in[2];
  const float* q_z_b    = (const float*)d_in[3];
  const float* q_feat_w = (const float*)d_in[4];
  const float* q_feat_b = (const float*)d_in[5];
  const float* q_gamma  = (const float*)d_in[6];
  const float* cq       = (const float*)d_in[7];
  float* out = (float*)d_out;

  (void)hipFuncSetAttribute(
      (const void*)PrimitiveAttentiveAtlasEncoder_85804856640021_kernel,
      hipFuncAttributeMaxDynamicSharedMemorySize, SMEM_BYTES);

  dim3 grid(B_TOT / ROWS_PER_BLOCK);   // 128 workgroups
  dim3 block(THREADS);                 // 8 wave32
  PrimitiveAttentiveAtlasEncoder_85804856640021_kernel<<<grid, block, SMEM_BYTES, stream>>>(
      z, features, q_z_w, q_z_b, q_feat_w, q_feat_b, q_gamma, cq, out);
}
